// Block_83159156785494
// MI455X (gfx1250) — compile-verified, matrix operand order checked by reading, not run on hardware
//
#include <hip/hip_runtime.h>
#include <hip/hip_bf16.h>
#include <math.h>

typedef __bf16 bf16_t;
typedef __attribute__((ext_vector_type(8)))  __bf16 v8bf;
typedef __attribute__((ext_vector_type(16))) __bf16 v16bf;
typedef __attribute__((ext_vector_type(8)))  float  v8f;
typedef __attribute__((ext_vector_type(4)))  int    v4i;

// ---------------------------------------------------------------------------
// Problem constants
// ---------------------------------------------------------------------------
#define BB   2
#define TT   2048
#define HH   2048
#define QHN  16
#define KHN  8
#define DD   128
#define FF   8192
#define MM   (BB * TT)          // 4096 rows

// ---------------------------------------------------------------------------
// Async global->LDS staging (CDNA5 GLOBAL_LOAD_ASYNC_TO_LDS_B128, ASYNCcnt).
// Builtin signature (from hipcc diagnostic): (v4i addrspace(1)*, v4i addrspace(3)*,
// imm offset, imm cpol). Falls back to synchronous uint4 copies if absent.
// ---------------------------------------------------------------------------
#define LDS_AS __attribute__((address_space(3)))
#define GLB_AS __attribute__((address_space(1)))

#if defined(__AMDGCN__) && \
    __has_builtin(__builtin_amdgcn_global_load_async_to_lds_b128) && \
    __has_builtin(__builtin_amdgcn_s_wait_asynccnt)
#define USE_ASYNC_LDS 1
#else
#define USE_ASYNC_LDS 0
#endif

__device__ __forceinline__ void async_copy16(const bf16_t* g, bf16_t* l) {
#if USE_ASYNC_LDS
    __builtin_amdgcn_global_load_async_to_lds_b128(
        (GLB_AS v4i*)(const void*)g, (LDS_AS v4i*)(void*)l, 0, 0);
#else
    *(uint4*)l = *(const uint4*)g;
#endif
}
__device__ __forceinline__ void async_wait() {
#if USE_ASYNC_LDS
    __builtin_amdgcn_s_wait_asynccnt(0);
#endif
}

// ---------------------------------------------------------------------------
// WMMA fragment loader.
// A (16-bit, 16x32) lane layout (ISA 7.12.2): lanes 0-15 hold M=0..15,
// VGPR0-3 = K 0..7, VGPR4-7 = K 16..23; lanes 16-31 hold K+8.
// Same pattern for B when B^T is stored n-major ([n][k]) in LDS.
// ---------------------------------------------------------------------------
__device__ __forceinline__ v16bf load_frag(const bf16_t* base, int lda, int lane) {
    int r  = lane & 15;
    int kh = (lane >> 4) << 3;               // 0 or 8
    const bf16_t* p = base + r * lda;
    v8bf lo = *(const v8bf*)(p + kh);        // K = kh .. kh+7
    v8bf hi = *(const v8bf*)(p + 16 + kh);   // K = 16+kh .. 23+kh
    return __builtin_shufflevector(lo, hi, 0,1,2,3,4,5,6,7,8,9,10,11,12,13,14,15);
}

__device__ __forceinline__ v8f wmma_bf16(v16bf a, v16bf b, v8f c) {
    return __builtin_amdgcn_wmma_f32_16x16x32_bf16(false, a, false, b,
                                                   (short)0, c, false, false);
}

// ---------------------------------------------------------------------------
// fp32 [K][N]  ->  bf16 [N][K]   (weight convert + transpose, LDS tiled)
// ---------------------------------------------------------------------------
__global__ __launch_bounds__(256)
void transpose_cvt_kernel(const float* __restrict__ src, bf16_t* __restrict__ dst,
                          int K, int N) {
    __shared__ float tile[32][33];
    int bn = blockIdx.x * 32;
    int bk = blockIdx.y * 32;
    int tx = threadIdx.x & 31;
    int ty = (threadIdx.x >> 5) * 4;
#pragma unroll
    for (int i = 0; i < 4; ++i)
        tile[ty + i][tx] = src[(size_t)(bk + ty + i) * N + bn + tx];
    __syncthreads();
#pragma unroll
    for (int i = 0; i < 4; ++i)
        dst[(size_t)(bn + ty + i) * K + bk + tx] = (bf16_t)tile[tx][ty + i];
}

// ---------------------------------------------------------------------------
// Row RMSNorm: fp32 in -> bf16 out (gamma fp32)
// ---------------------------------------------------------------------------
__global__ __launch_bounds__(256)
void rmsnorm_kernel(const float* __restrict__ x, const float* __restrict__ gamma,
                    bf16_t* __restrict__ out, int Hdim) {
    int row = blockIdx.x;
    const float* xr = x + (size_t)row * Hdim;
    float ss = 0.f;
    for (int i = threadIdx.x; i < Hdim; i += 256) { float v = xr[i]; ss += v * v; }
#pragma unroll
    for (int o = 16; o > 0; o >>= 1) ss += __shfl_xor(ss, o, 32);
    __shared__ float red[8];
    int wave = threadIdx.x >> 5, lane = threadIdx.x & 31;
    if (lane == 0) red[wave] = ss;
    __syncthreads();
    if (wave == 0) {
        float v = (lane < 8) ? red[lane] : 0.f;
#pragma unroll
        for (int o = 4; o > 0; o >>= 1) v += __shfl_xor(v, o, 32);
        if (lane == 0) red[0] = v;
    }
    __syncthreads();
    float inv = 1.f / sqrtf(red[0] / (float)Hdim + 1e-6f);
    bf16_t* orow = out + (size_t)row * Hdim;
    for (int i = threadIdx.x; i < Hdim; i += 256)
        orow[i] = (bf16_t)(gamma[i] * xr[i] * inv);
}

// ---------------------------------------------------------------------------
// Per-head RMSNorm + RoPE, in place on bf16 rows; optional fp32 copy (k output)
// One block (128 threads) per (token, head).
// ---------------------------------------------------------------------------
__global__ __launch_bounds__(128)
void rope_kernel(bf16_t* __restrict__ qk, const float* __restrict__ gamma,
                 const float* __restrict__ sinp, const float* __restrict__ cosp,
                 float* __restrict__ out_f32, int heads) {
    int idx = blockIdx.x;
    int h = idx % heads;
    int m = idx / heads;
    int d = threadIdx.x;                      // 0..127
    bf16_t* row = qk + (size_t)m * heads * DD + (size_t)h * DD;
    float v = (float)row[d];
    float ss = v * v;
#pragma unroll
    for (int o = 16; o > 0; o >>= 1) ss += __shfl_xor(ss, o, 32);
    __shared__ float sred[4];
    __shared__ float vals[DD];
    if ((threadIdx.x & 31) == 0) sred[threadIdx.x >> 5] = ss;
    __syncthreads();
    float rms = sqrtf((sred[0] + sred[1] + sred[2] + sred[3]) / (float)DD + 1e-6f);
    float xn = (float)(bf16_t)(gamma[d] * v / rms);   // rms_norm rounds to bf16
    vals[d] = xn;
    __syncthreads();
    const float* s = sinp + (size_t)m * (DD / 2);
    const float* c = cosp + (size_t)m * (DD / 2);
    float o;
    if (d < DD / 2) o = vals[d] * c[d] - vals[d + DD / 2] * s[d];
    else            o = vals[d] * c[d - DD / 2] + vals[d - DD / 2] * s[d - DD / 2];
    row[d] = (bf16_t)o;
    if (out_f32) out_f32[(size_t)m * heads * DD + (size_t)h * DD + d] = o;
}

// ---------------------------------------------------------------------------
// Tiled WMMA GEMM: C[MxN] = A_bf16[MxK] * Bt_bf16[NxK]^T
// Block 128x128x32, 256 threads = 8 waves (4 along M x 2 along N),
// each wave computes a 32x64 patch = 2x4 WMMA accumulators.
// Double-buffered LDS; async global->LDS staging overlapped with WMMA.
// MODE 0: store bf16
// MODE 1: out_f32 = resid + C
// MODE 2: dual-B SwiGLU: out_bf16 = silu(A*Bt) * (A*Bt2)
// ---------------------------------------------------------------------------
#define GBM 128
#define GBN 128
#define GBK 32
#define GLD 40   // padded LDS row stride (bf16 elems)

template<int MODE>
__global__ __launch_bounds__(256)
void gemm_wmma_kernel(const bf16_t* __restrict__ A, const bf16_t* __restrict__ Bt,
                      const bf16_t* __restrict__ Bt2, const float* __restrict__ resid,
                      void* __restrict__ outp, int M, int N, int K) {
    __shared__ bf16_t As[2][GBM][GLD];
    __shared__ bf16_t Bs[2][GBN][GLD];
    __shared__ bf16_t Bs2[(MODE == 2) ? 2 : 1][(MODE == 2) ? GBN : 1][(MODE == 2) ? GLD : 8];

    int tid  = threadIdx.x;
    int lane = tid & 31;
    int wave = tid >> 5;
    int wm = (wave & 3) * 32;
    int wn = (wave >> 2) * 64;
    int m0 = blockIdx.y * GBM;
    int n0 = blockIdx.x * GBN;

    v8f acc[2][4]  = {};
    v8f acc2[2][4] = {};

    auto stage = [&](int k0, int buf) {
#pragma unroll
        for (int p = 0; p < 2; ++p) {
            int idx = p * 256 + tid;
            int r = idx >> 2;
            int c = (idx & 3) * 8;
            async_copy16(A  + (size_t)(m0 + r) * K + k0 + c, &As[buf][r][c]);
            async_copy16(Bt + (size_t)(n0 + r) * K + k0 + c, &Bs[buf][r][c]);
            if constexpr (MODE == 2)
                async_copy16(Bt2 + (size_t)(n0 + r) * K + k0 + c, &Bs2[buf][r][c]);
        }
    };

    int nk = K / GBK;
    stage(0, 0);

    for (int kt = 0; kt < nk; ++kt) {
        int buf = kt & 1;
        async_wait();            // stage(kt) complete (this wave's async ops)
        __syncthreads();         // all waves' tiles visible
        if (kt + 1 < nk) stage((kt + 1) * GBK, buf ^ 1);   // in flight over compute

        v16bf af0 = load_frag(&As[buf][wm][0],      GLD, lane);
        v16bf af1 = load_frag(&As[buf][wm + 16][0], GLD, lane);
        v16bf bfr[4];
#pragma unroll
        for (int ni = 0; ni < 4; ++ni)
            bfr[ni] = load_frag(&Bs[buf][wn + ni * 16][0], GLD, lane);
#pragma unroll
        for (int ni = 0; ni < 4; ++ni) {
            acc[0][ni] = wmma_bf16(af0, bfr[ni], acc[0][ni]);
            acc[1][ni] = wmma_bf16(af1, bfr[ni], acc[1][ni]);
        }
        if constexpr (MODE == 2) {
#pragma unroll
            for (int ni = 0; ni < 4; ++ni) {
                v16bf bf2 = load_frag(&Bs2[buf][wn + ni * 16][0], GLD, lane);
                acc2[0][ni] = wmma_bf16(af0, bf2, acc2[0][ni]);
                acc2[1][ni] = wmma_bf16(af1, bf2, acc2[1][ni]);
            }
        }
    }

    // Epilogue. C layout (ISA 7.12.2): lanes 0-15 = N cols, VGPR j = row j;
    // lanes 16-31 = rows j+8.
    int half = (lane >> 4) ? 8 : 0;
#pragma unroll
    for (int mi = 0; mi < 2; ++mi)
#pragma unroll
        for (int ni = 0; ni < 4; ++ni)
#pragma unroll
            for (int j = 0; j < 8; ++j) {
                int m = m0 + wm + mi * 16 + j + half;
                int n = n0 + wn + ni * 16 + (lane & 15);
                size_t o = (size_t)m * N + n;
                float cv = acc[mi][ni][j];
                if constexpr (MODE == 0) {
                    ((bf16_t*)outp)[o] = (bf16_t)cv;
                } else if constexpr (MODE == 1) {
                    ((float*)outp)[o] = resid[o] + cv;
                } else {
                    float g = cv;
                    float u = acc2[mi][ni][j];
                    float sg = g / (1.f + __expf(-g));
                    ((bf16_t*)outp)[o] = (bf16_t)(sg * u);
                }
            }
}

// ---------------------------------------------------------------------------
// Flash attention (causal, GQA 16Q/8KV, D=128), bf16 WMMA for QK^T and PV.
// 1 block = (b, qh, 64-row q tile); 4 waves, each owns 16 q rows.
// ---------------------------------------------------------------------------
#define AQ  64
#define AK  64
#define LQK 136   // padded stride for Qs/Ks (bf16)
#define LV  72    // padded stride for Vs/Ps (bf16)

__global__ __launch_bounds__(128)
void attn_kernel(const bf16_t* __restrict__ q, const bf16_t* __restrict__ k,
                 const bf16_t* __restrict__ v, const int* __restrict__ tmask,
                 bf16_t* __restrict__ out) {
    __shared__ bf16_t Qs[AQ][LQK];
    __shared__ bf16_t Ks[AK][LQK];
    __shared__ bf16_t Vs[DD][LV];      // V transposed: [d][token]
    __shared__ bf16_t Ps[AQ][LV];      // per-wave P staging

    int qt  = blockIdx.x;              // q tile (0..31)
    int qh  = blockIdx.y;              // q head
    int b   = blockIdx.z;
    int kvh = qh >> 1;                 // GQA mapping (rep=2)
    int tid = threadIdx.x, lane = tid & 31, wave = tid >> 5;
    int half = (lane >> 4) ? 8 : 0;

    // Load Q tile: 64 x 128 bf16
#pragma unroll
    for (int p = 0; p < 8; ++p) {
        int idx = p * 128 + tid;
        int r = idx >> 4, c = (idx & 15) * 8;
        *(uint4*)&Qs[r][c] =
            *(const uint4*)(q + (size_t)(b * TT + qt * AQ + r) * (QHN * DD) + qh * DD + c);
    }

    float mstate[8], lstate[8];
    v8f oacc[8] = {};
#pragma unroll
    for (int j = 0; j < 8; ++j) { mstate[j] = -1e30f; lstate[j] = 0.f; }

    const float scale = 0.08838834764831845f;   // 1/sqrt(128)
    const int* tm = tmask + b * TT;

    for (int kb = 0; kb <= qt; ++kb) {
        // Stage K (async, row-major) and V (transposed through registers)
#pragma unroll
        for (int p = 0; p < 8; ++p) {
            int idx = p * 128 + tid;
            int r = idx >> 4, c = (idx & 15) * 8;
            const bf16_t* kvrow =
                k + (size_t)(b * TT + kb * AK + r) * (KHN * DD) + kvh * DD + c;
            async_copy16(kvrow, &Ks[r][c]);
            v8bf vv = *(const v8bf*)(v + (size_t)(b * TT + kb * AK + r) * (KHN * DD) + kvh * DD + c);
#pragma unroll
            for (int i = 0; i < 8; ++i) Vs[c + i][r] = vv[i];
        }
        if (kb + 1 <= qt)   // hint next K tile toward L2
            __builtin_prefetch(k + (size_t)(b * TT + (kb + 1) * AK + (tid >> 1)) * (KHN * DD) + kvh * DD, 0, 1);
        async_wait();
        __syncthreads();

        // S = Q K^T  (16 q rows x 64 k cols per wave)
        v8f sacc[4] = {};
#pragma unroll
        for (int kd = 0; kd < 4; ++kd) {
            v16bf aq = load_frag(&Qs[wave * 16][kd * 32], LQK, lane);
            v16bf bk[4];
#pragma unroll
            for (int nt = 0; nt < 4; ++nt)
                bk[nt] = load_frag(&Ks[nt * 16][kd * 32], LQK, lane);
#pragma unroll
            for (int nt = 0; nt < 4; ++nt)
                sacc[nt] = wmma_bf16(aq, bk[nt], sacc[nt]);
        }

        // scale + causal/token mask
#pragma unroll
        for (int nt = 0; nt < 4; ++nt) {
            int kcol = kb * AK + nt * 16 + (lane & 15);
            bool kok = tm[kcol] != 0;
#pragma unroll
            for (int j = 0; j < 8; ++j) {
                int qrow = qt * AQ + wave * 16 + j + half;
                float sv = sacc[nt][j] * scale;
                if (!kok || kcol > qrow) sv = -1e30f;
                sacc[nt][j] = sv;
            }
        }

        // online softmax per row (16-lane xor reductions stay within halves)
#pragma unroll
        for (int j = 0; j < 8; ++j) {
            float mx = fmaxf(fmaxf(sacc[0][j], sacc[1][j]), fmaxf(sacc[2][j], sacc[3][j]));
#pragma unroll
            for (int o = 8; o > 0; o >>= 1) mx = fmaxf(mx, __shfl_xor(mx, o, 32));
            float mnew = fmaxf(mstate[j], mx);
            float corr = __expf(mstate[j] - mnew);
            mstate[j] = mnew;
            float rs = 0.f;
#pragma unroll
            for (int nt = 0; nt < 4; ++nt) {
                float p = __expf(sacc[nt][j] - mnew);
                sacc[nt][j] = p;
                rs += p;
            }
#pragma unroll
            for (int o = 8; o > 0; o >>= 1) rs += __shfl_xor(rs, o, 32);
            lstate[j] = lstate[j] * corr + rs;
#pragma unroll
            for (int dt = 0; dt < 8; ++dt) oacc[dt][j] *= corr;
            int prow = wave * 16 + j + half;
#pragma unroll
            for (int nt = 0; nt < 4; ++nt)
                Ps[prow][nt * 16 + (lane & 15)] = (bf16_t)sacc[nt][j];
        }

        // O += P V   (P per-wave in LDS; A-frag repack)
#pragma unroll
        for (int kc = 0; kc < 2; ++kc) {
            v16bf ap = load_frag(&Ps[wave * 16][kc * 32], LV, lane);
#pragma unroll
            for (int dt = 0; dt < 8; ++dt) {
                v16bf bv = load_frag(&Vs[dt * 16][kc * 32], LV, lane);
                oacc[dt] = wmma_bf16(ap, bv, oacc[dt]);
            }
        }
        __syncthreads();
    }

    // normalize and write attn output (bf16, [m][QH*D])
#pragma unroll
    for (int dt = 0; dt < 8; ++dt)
#pragma unroll
        for (int j = 0; j < 8; ++j) {
            int qrow = qt * AQ + wave * 16 + j + half;
            int d = dt * 16 + (lane & 15);
            out[(size_t)(b * TT + qrow) * (QHN * DD) + qh * DD + d] =
                (bf16_t)(oacc[dt][j] / lstate[j]);
        }
}

// ---------------------------------------------------------------------------
// Simple bf16 copy (v -> d_out region)
// ---------------------------------------------------------------------------
__global__ __launch_bounds__(256)
void copy_bf16_kernel(const bf16_t* __restrict__ src, bf16_t* __restrict__ dst, int n) {
    int i = blockIdx.x * 256 + threadIdx.x;
    if (i < n) dst[i] = src[i];
}

// ---------------------------------------------------------------------------
// Host-side orchestration
// ---------------------------------------------------------------------------
extern "C" void kernel_launch(void* const* d_in, const int* in_sizes, int n_in,
                              void* d_out, int out_size, void* d_ws, size_t ws_size,
                              hipStream_t stream) {
    (void)in_sizes; (void)n_in; (void)out_size; (void)ws_size;

    const float* x      = (const float*)d_in[0];
    const float* sinp   = (const float*)d_in[1];
    const float* cosp   = (const float*)d_in[2];
    const int*   tmask  = (const int*)d_in[3];
    /* d_in[4] = layer_id (unused) */
    const float* pre_g  = (const float*)d_in[5];
    const float* wq     = (const float*)d_in[6];
    const float* wk     = (const float*)d_in[7];
    const float* wv     = (const float*)d_in[8];
    const float* qg     = (const float*)d_in[9];
    const float* kg     = (const float*)d_in[10];
    const float* wo     = (const float*)d_in[11];
    const float* post_g = (const float*)d_in[12];
    const float* wg     = (const float*)d_in[13];
    const float* wu     = (const float*)d_in[14];
    const float* wd     = (const float*)d_in[15];

    // d_out: x (f32) | k (f32) | v (bf16), concatenated in return order
    float*  out_x = (float*)d_out;
    float*  out_k = out_x + (size_t)MM * HH;
    bf16_t* out_v = (bf16_t*)((char*)d_out +
                              ((size_t)MM * HH + (size_t)MM * KHN * DD) * sizeof(float));

    // workspace carve-up
    char* ws = (char*)d_ws;
    size_t off = 0;
    auto alloc = [&](size_t bytes) {
        size_t o = off;
        off += (bytes + 255) & ~(size_t)255;
        return o;
    };
    bf16_t* xnorm  = (bf16_t*)(ws + alloc((size_t)MM * HH * 2));
    bf16_t* wq_t   = (bf16_t*)(ws + alloc((size_t)HH * (QHN * DD) * 2));
    bf16_t* wk_t   = (bf16_t*)(ws + alloc((size_t)HH * (KHN * DD) * 2));
    bf16_t* wv_t   = (bf16_t*)(ws + alloc((size_t)HH * (KHN * DD) * 2));
    bf16_t* wo_t   = (bf16_t*)(ws + alloc((size_t)(QHN * DD) * HH * 2));
    bf16_t* wg_t   = (bf16_t*)(ws + alloc((size_t)HH * FF * 2));
    bf16_t* wu_t   = (bf16_t*)(ws + alloc((size_t)HH * FF * 2));
    bf16_t* wd_t   = (bf16_t*)(ws + alloc((size_t)FF * HH * 2));
    bf16_t* qb     = (bf16_t*)(ws + alloc((size_t)MM * (QHN * DD) * 2));
    bf16_t* kb     = (bf16_t*)(ws + alloc((size_t)MM * (KHN * DD) * 2));
    bf16_t* vb     = (bf16_t*)(ws + alloc((size_t)MM * (KHN * DD) * 2));
    bf16_t* attn   = (bf16_t*)(ws + alloc((size_t)MM * (QHN * DD) * 2));
    float*  x1     = (float*) (ws + alloc((size_t)MM * HH * 4));
    bf16_t* xnorm2 = (bf16_t*)(ws + alloc((size_t)MM * HH * 2));
    bf16_t* yb     = (bf16_t*)(ws + alloc((size_t)MM * FF * 2));

    // 1. weight convert + transpose (fp32 [K][N] -> bf16 [N][K])
    transpose_cvt_kernel<<<dim3(QHN*DD/32, HH/32), 256, 0, stream>>>(wq, wq_t, HH, QHN*DD);
    transpose_cvt_kernel<<<dim3(KHN*DD/32, HH/32), 256, 0, stream>>>(wk, wk_t, HH, KHN*DD);
    transpose_cvt_kernel<<<dim3(KHN*DD/32, HH/32), 256, 0, stream>>>(wv, wv_t, HH, KHN*DD);
    transpose_cvt_kernel<<<dim3(HH/32, QHN*DD/32), 256, 0, stream>>>(wo, wo_t, QHN*DD, HH);
    transpose_cvt_kernel<<<dim3(FF/32, HH/32),     256, 0, stream>>>(wg, wg_t, HH, FF);
    transpose_cvt_kernel<<<dim3(FF/32, HH/32),     256, 0, stream>>>(wu, wu_t, HH, FF);
    transpose_cvt_kernel<<<dim3(HH/32, FF/32),     256, 0, stream>>>(wd, wd_t, FF, HH);

    // 2. pre-RMSNorm
    rmsnorm_kernel<<<MM, 256, 0, stream>>>(x, pre_g, xnorm, HH);

    // 3. QKV projections (bf16 WMMA GEMMs)
    gemm_wmma_kernel<0><<<dim3((QHN*DD)/GBN, MM/GBM), 256, 0, stream>>>(
        xnorm, wq_t, nullptr, nullptr, qb, MM, QHN*DD, HH);
    gemm_wmma_kernel<0><<<dim3((KHN*DD)/GBN, MM/GBM), 256, 0, stream>>>(
        xnorm, wk_t, nullptr, nullptr, kb, MM, KHN*DD, HH);
    gemm_wmma_kernel<0><<<dim3((KHN*DD)/GBN, MM/GBM), 256, 0, stream>>>(
        xnorm, wv_t, nullptr, nullptr, vb, MM, KHN*DD, HH);

    // 4. head RMSNorm + RoPE (k also written fp32 to d_out)
    rope_kernel<<<MM * QHN, 128, 0, stream>>>(qb, qg, sinp, cosp, nullptr, QHN);
    rope_kernel<<<MM * KHN, 128, 0, stream>>>(kb, kg, sinp, cosp, out_k, KHN);

    // 5. v output copy
    copy_bf16_kernel<<<(MM * KHN * DD + 255) / 256, 256, 0, stream>>>(
        vb, out_v, MM * KHN * DD);

    // 6. causal GQA flash attention
    attn_kernel<<<dim3(TT / AQ, QHN, BB), 128, 0, stream>>>(qb, kb, vb, tmask, attn);

    // 7. output projection + residual -> x1 (f32)
    gemm_wmma_kernel<1><<<dim3(HH/GBN, MM/GBM), 256, 0, stream>>>(
        attn, wo_t, nullptr, x, x1, MM, HH, QHN*DD);

    // 8. post-RMSNorm
    rmsnorm_kernel<<<MM, 256, 0, stream>>>(x1, post_g, xnorm2, HH);

    // 9. fused SwiGLU (gate + up in one pass) -> y (bf16)
    gemm_wmma_kernel<2><<<dim3(FF/GBN, MM/GBM), 256, 0, stream>>>(
        xnorm2, wg_t, wu_t, nullptr, yb, MM, FF, HH);

    // 10. down projection + residual -> final x (f32, d_out)
    gemm_wmma_kernel<1><<<dim3(HH/GBN, MM/GBM), 256, 0, stream>>>(
        yb, wd_t, nullptr, x1, out_x, MM, HH, FF);
}
